// GraphBlock_37048387895511
// MI455X (gfx1250) — matrix operand order repaired and checked by reading, not compiled, not run on hardware
//
#include <hip/hip_runtime.h>

#define DEV __device__ __forceinline__

typedef __attribute__((ext_vector_type(16))) __bf16 bf16x16;
typedef __attribute__((ext_vector_type(8)))  float  floatx8;

constexpr int NSEQ = 512;
constexpr int DDIM = 128;
constexpr int EDIM = 64;
constexpr int HNUM = 8;
constexpr int HDIM = 32;
constexpr int EP   = NSEQ * NSEQ;   // 262144 edge positions
constexpr float LN_EPS = 1e-5f;

// ---------------------------------------------------------------- WMMA helpers
DEV floatx8 wmma_bf16(bf16x16 a, bf16x16 b, floatx8 c) {
  return __builtin_amdgcn_wmma_f32_16x16x32_bf16(false, a, false, b, (short)0, c,
                                                 false, false);
}

// A fragment: 16x32 bf16 tile, row-major source with leading dim lda.
// lane<16: M=lane, K slots {k0+h, k0+16+h}; lane>=16: K base +8 (ISA 7.12.2).
DEV bf16x16 load_a_frag(const __bf16* A, int lda, int m0, int k0, int lane) {
  const __bf16* p = A + (size_t)(m0 + (lane & 15)) * lda + k0 + ((lane >> 4) << 3);
  bf16x16 a;
#pragma unroll
  for (int h = 0; h < 8; ++h) { a[h] = p[h]; a[h + 8] = p[h + 16]; }
  return a;
}

// B fragment: 32x16 bf16 tile, N = lane&15, K slots {k0+h, k0+16+h}; lane>=16: K base +8.
DEV bf16x16 load_b_frag(const __bf16* B, int ldb, int k0, int n0, int lane) {
  const __bf16* p = B + (size_t)(k0 + ((lane >> 4) << 3)) * ldb + n0 + (lane & 15);
  bf16x16 b;
#pragma unroll
  for (int h = 0; h < 8; ++h) { b[h] = p[(size_t)h * ldb]; b[h + 8] = p[(size_t)(h + 16) * ldb]; }
  return b;
}

DEV float sigmoidf_(float x) { return 1.f / (1.f + __expf(-x)); }

// ---------------------------------------------------------------- pack weights
__global__ void k_pack(const float* __restrict__ src, __bf16* __restrict__ dst, int n) {
  int i = blockIdx.x * blockDim.x + threadIdx.x;
  if (i < n) dst[i] = (__bf16)src[i];
}

// ---------------------------------------------------------------- row LayerNorm
// one wave per row; C in {64,128}; bf16 output feeds WMMA A operands.
__global__ void k_ln(const float* __restrict__ x, const float* __restrict__ g,
                     const float* __restrict__ b, __bf16* __restrict__ outh, int C) {
  int lane = threadIdx.x & 31;
  size_t row = (size_t)blockIdx.x * (blockDim.x >> 5) + (threadIdx.x >> 5);
  const float* xr = x + row * C;
  float s = 0.f, s2 = 0.f;
  for (int c = lane; c < C; c += 32) { float v = xr[c]; s += v; s2 += v * v; }
#pragma unroll
  for (int o = 16; o > 0; o >>= 1) { s += __shfl_xor(s, o, 32); s2 += __shfl_xor(s2, o, 32); }
  float mean = s / C;
  float rstd = rsqrtf(s2 / C - mean * mean + LN_EPS);
  __bf16* orow = outh + row * C;
  for (int c = lane; c < C; c += 32)
    orow[c] = (__bf16)((xr[c] - mean) * rstd * g[c] + b[c]);
}

// ------------------------------------------- generic GEMM: C = act(A*B + bias) + res
// A bf16 [M,K] (lda=K), B bf16 [K,N] (ldb=N). grid=(M/16, N/64), block=32.
__global__ void k_gemm(const __bf16* __restrict__ A, const __bf16* __restrict__ B,
                       const float* __restrict__ bias, const float* __restrict__ residual,
                       float* __restrict__ outf, __bf16* __restrict__ outh,
                       int N, int K, int act) {
  int lane = threadIdx.x;
  int m0 = blockIdx.x * 16, n0 = blockIdx.y * 64;
  floatx8 acc[4] = {};
  for (int k0 = 0; k0 < K; k0 += 32) {
    bf16x16 a = load_a_frag(A, K, m0, k0, lane);
#pragma unroll
    for (int s = 0; s < 4; ++s) {
      bf16x16 b = load_b_frag(B, N, k0, n0 + s * 16, lane);
      acc[s] = wmma_bf16(a, b, acc[s]);
    }
  }
  int col = lane & 15, rowhi = (lane >> 4) << 3;
#pragma unroll
  for (int s = 0; s < 4; ++s)
#pragma unroll
    for (int r = 0; r < 8; ++r) {
      int m = m0 + rowhi + r, n = n0 + s * 16 + col;
      float v = acc[s][r];
      if (bias) v += bias[n];
      if (act == 1) v = fmaxf(v, 0.f);
      else if (act == 2) v = sigmoidf_(v);
      size_t off = (size_t)m * N + n;
      if (residual) v += residual[off];
      if (outf) outf[off] = v;
      if (outh) outh[off] = (__bf16)v;
    }
}

// --------------------------------- attention bias: bias[h,i,j] = ln(edge)[i,j,:]@wb
__global__ void k_bias(const float* __restrict__ edge, const float* __restrict__ g,
                       const float* __restrict__ b, const float* __restrict__ wb,
                       float* __restrict__ bias) {
  int lane = threadIdx.x & 31;
  size_t p = (size_t)blockIdx.x * (blockDim.x >> 5) + (threadIdx.x >> 5);
  const float* x = edge + p * 64;
  float v0 = x[lane], v1 = x[lane + 32];
  float s = v0 + v1, s2 = v0 * v0 + v1 * v1;
#pragma unroll
  for (int o = 16; o > 0; o >>= 1) { s += __shfl_xor(s, o, 32); s2 += __shfl_xor(s2, o, 32); }
  float mean = s * (1.f / 64.f);
  float rstd = rsqrtf(s2 * (1.f / 64.f) - mean * mean + LN_EPS);
  float y0 = (v0 - mean) * rstd * g[lane] + b[lane];
  float y1 = (v1 - mean) * rstd * g[lane + 32] + b[lane + 32];
  float dot[8];
#pragma unroll
  for (int h = 0; h < 8; ++h)
    dot[h] = y0 * wb[lane * 8 + h] + y1 * wb[(lane + 32) * 8 + h];
#pragma unroll
  for (int h = 0; h < 8; ++h)
#pragma unroll
    for (int o = 16; o > 0; o >>= 1) dot[h] += __shfl_xor(dot[h], o, 32);
  if (lane < 8) bias[(size_t)lane * EP + p] = dot[lane];
}

// ------------------------- kT[h][d][j] = k[j][h*32+d]  (B operand for logits GEMM)
__global__ void k_transpose_k(const __bf16* __restrict__ kb, __bf16* __restrict__ kT) {
  int idx = blockIdx.x * blockDim.x + threadIdx.x;   // over 512*256
  int j = idx >> 8, hd = idx & 255;
  kT[(size_t)hd * 512 + j] = kb[idx];
}

// ---------------- logits[h,i,j] = scale * q_h @ kT_h + bias (in place over bias buf)
__global__ void k_logits(const __bf16* __restrict__ qb, const __bf16* __restrict__ kT,
                         float* __restrict__ logits) {
  int lane = threadIdx.x;
  int i0 = blockIdx.x * 16, j0 = blockIdx.y * 64, h = blockIdx.z;
  bf16x16 a = load_a_frag(qb + h * 32, 256, i0, 0, lane);   // K = 32 = one step
  floatx8 acc[4] = {};
#pragma unroll
  for (int s = 0; s < 4; ++s) {
    bf16x16 b = load_b_frag(kT + (size_t)h * 32 * 512, 512, 0, j0 + s * 16, lane);
    acc[s] = wmma_bf16(a, b, acc[s]);
  }
  const float scale = 0.17677669529663687f;  // 32^-0.5
  int col = lane & 15, rowhi = (lane >> 4) << 3;
#pragma unroll
  for (int s = 0; s < 4; ++s)
#pragma unroll
    for (int r = 0; r < 8; ++r) {
      size_t off = (size_t)h * EP + (size_t)(i0 + rowhi + r) * 512 + j0 + s * 16 + col;
      logits[off] = acc[s][r] * scale + logits[off];
    }
}

// ----------------------------------------------- softmax rows (h,i) -> bf16 attn
__global__ void k_softmax(const float* __restrict__ logits, __bf16* __restrict__ out) {
  int lane = threadIdx.x & 31;
  size_t row = (size_t)blockIdx.x * (blockDim.x >> 5) + (threadIdx.x >> 5);
  const float* p = logits + row * 512;
  float mx = -1e30f;
#pragma unroll
  for (int t = 0; t < 16; ++t) mx = fmaxf(mx, p[lane + 32 * t]);
#pragma unroll
  for (int o = 16; o > 0; o >>= 1) mx = fmaxf(mx, __shfl_xor(mx, o, 32));
  float e[16], sum = 0.f;
#pragma unroll
  for (int t = 0; t < 16; ++t) { e[t] = __expf(p[lane + 32 * t] - mx); sum += e[t]; }
#pragma unroll
  for (int o = 16; o > 0; o >>= 1) sum += __shfl_xor(sum, o, 32);
  float inv = 1.f / sum;
  __bf16* q = out + row * 512;
#pragma unroll
  for (int t = 0; t < 16; ++t) q[lane + 32 * t] = (__bf16)(e[t] * inv);
}

// -------------------------- res[i,h*32+d] = (attn_h @ v_h) * g ; bf16 out for GEMM
__global__ void k_av(const __bf16* __restrict__ attnb, const __bf16* __restrict__ vb,
                     const float* __restrict__ gbuf, __bf16* __restrict__ resg) {
  int lane = threadIdx.x;
  int i0 = blockIdx.x * 16, h = blockIdx.y;
  const __bf16* A = attnb + (size_t)h * EP;
  floatx8 acc[2] = {};
  for (int k0 = 0; k0 < 512; k0 += 32) {
    bf16x16 a = load_a_frag(A, 512, i0, k0, lane);
#pragma unroll
    for (int s = 0; s < 2; ++s) {
      bf16x16 b = load_b_frag(vb + h * 32, 256, k0, s * 16, lane);
      acc[s] = wmma_bf16(a, b, acc[s]);
    }
  }
  int col = lane & 15, rowhi = (lane >> 4) << 3;
#pragma unroll
  for (int s = 0; s < 2; ++s)
#pragma unroll
    for (int r = 0; r < 8; ++r) {
      size_t off = (size_t)(i0 + rowhi + r) * 256 + h * 32 + s * 16 + col;
      resg[off] = (__bf16)(acc[s][r] * gbuf[off]);
    }
}

// ------- NodeToEdge fused: A rows generated on the fly from s=[q2|k2]; +edge resid
__global__ void k_n2e(const float* __restrict__ s, const __bf16* __restrict__ W,
                      const float* __restrict__ bo, const float* __restrict__ edge_in,
                      float* __restrict__ edge_out) {
  int lane = threadIdx.x;
  int i = blockIdx.y, j0 = blockIdx.x * 16;
  int jr = j0 + (lane & 15);
  int base = (lane >> 4) << 3;
  const float* qrow = s + (size_t)jr * 256;        // q2[j,d]
  const float* krow = s + (size_t)i * 256 + 128;   // k2[i,d]
  floatx8 acc[4] = {};
  for (int kb = 0; kb < 256; kb += 32) {
    bf16x16 a;
#pragma unroll
    for (int h = 0; h < 16; ++h) {
      int kl = kb + base + (h < 8 ? h : h + 8);
      float v = (kl < 128) ? qrow[kl] * krow[kl] : qrow[kl - 128] - krow[kl - 128];
      a[h] = (__bf16)v;
    }
#pragma unroll
    for (int s4 = 0; s4 < 4; ++s4) {
      bf16x16 b = load_b_frag(W, 64, kb, s4 * 16, lane);
      acc[s4] = wmma_bf16(a, b, acc[s4]);
    }
  }
  int col = lane & 15, rowhi = (lane >> 4) << 3;
#pragma unroll
  for (int s4 = 0; s4 < 4; ++s4)
#pragma unroll
    for (int r = 0; r < 8; ++r) {
      int j = j0 + rowhi + r, n = s4 * 16 + col;
      size_t off = ((size_t)i * 512 + j) * 64 + n;
      edge_out[off] = edge_in[off] + bo[n] + acc[s4][r];
    }
}

// ---- gated projection GEMM: dst = (A@Wp+bp)*sigmoid(A@Wg+bg), scatter to d-planes
// transposed=0: dst[d*EP + p] ; transposed=1: dst[d*EP + (p%512)*512 + p/512]
__global__ void k_gated_gemm(const __bf16* __restrict__ A, const __bf16* __restrict__ Wp,
                             const float* __restrict__ bp, const __bf16* __restrict__ Wg,
                             const float* __restrict__ bg, __bf16* __restrict__ dst,
                             int transposed) {
  int lane = threadIdx.x;
  int m0 = blockIdx.x * 16;
  floatx8 ap[4] = {}, ag[4] = {};
  for (int k0 = 0; k0 < 64; k0 += 32) {
    bf16x16 a = load_a_frag(A, 64, m0, k0, lane);
#pragma unroll
    for (int s = 0; s < 4; ++s) {
      bf16x16 b1 = load_b_frag(Wp, 64, k0, s * 16, lane);
      ap[s] = wmma_bf16(a, b1, ap[s]);
      bf16x16 b2 = load_b_frag(Wg, 64, k0, s * 16, lane);
      ag[s] = wmma_bf16(a, b2, ag[s]);
    }
  }
  int col = lane & 15, rowhi = (lane >> 4) << 3;
#pragma unroll
  for (int s = 0; s < 4; ++s)
#pragma unroll
    for (int r = 0; r < 8; ++r) {
      int m = m0 + rowhi + r, d = s * 16 + col;
      float v = (ap[s][r] + bp[d]) * sigmoidf_(ag[s][r] + bg[d]);
      size_t off = (size_t)d * EP +
                   (transposed ? ((size_t)(m & 511) * 512 + (m >> 9)) : (size_t)m);
      dst[off] = (__bf16)v;
    }
}

// ------------------ batched (over d) 512x512x512 GEMM: C_d = A_d @ B_d, bf16 planes
__global__ void k_tri_mm(const __bf16* __restrict__ Abuf, const __bf16* __restrict__ Bbuf,
                         __bf16* __restrict__ Cbuf) {
  int lane = threadIdx.x;
  int i0 = blockIdx.x * 32, j0 = blockIdx.y * 64, d = blockIdx.z;
  const __bf16* Ad = Abuf + (size_t)d * EP;
  const __bf16* Bd = Bbuf + (size_t)d * EP;
  __bf16* Cd = Cbuf + (size_t)d * EP;
  floatx8 acc0[4] = {}, acc1[4] = {};
  for (int k0 = 0; k0 < 512; k0 += 32) {
    bf16x16 a0 = load_a_frag(Ad, 512, i0, k0, lane);
    bf16x16 a1 = load_a_frag(Ad, 512, i0 + 16, k0, lane);
#pragma unroll
    for (int s = 0; s < 4; ++s) {
      bf16x16 b = load_b_frag(Bd, 512, k0, j0 + s * 16, lane);
      acc0[s] = wmma_bf16(a0, b, acc0[s]);
      acc1[s] = wmma_bf16(a1, b, acc1[s]);
    }
  }
  int col = lane & 15, rowhi = (lane >> 4) << 3;
#pragma unroll
  for (int s = 0; s < 4; ++s)
#pragma unroll
    for (int r = 0; r < 8; ++r) {
      int j = j0 + s * 16 + col;
      Cd[(size_t)(i0 + rowhi + r) * 512 + j] = (__bf16)acc0[s][r];
      Cd[(size_t)(i0 + 16 + rowhi + r) * 512 + j] = (__bf16)acc1[s][r];
    }
}

// ---- tri post: LN over d + out-gate + out-linear (64x64) + residual into edge_cur
__global__ void k_tri_post(const __bf16* __restrict__ mm, const float* __restrict__ on_g,
                           const float* __restrict__ on_b, const __bf16* __restrict__ ogate,
                           const __bf16* __restrict__ Wo, const float* __restrict__ bo,
                           float* __restrict__ edge) {
  int lane = threadIdx.x;
  int p0 = blockIdx.x * 16;
  int p = p0 + (lane & 15);
  float vals[64];
  float s = 0.f, s2 = 0.f;
#pragma unroll
  for (int d = 0; d < 64; ++d) {
    float v = (float)mm[(size_t)d * EP + p];
    vals[d] = v; s += v; s2 += v * v;
  }
  float mean = s * (1.f / 64.f);
  float rstd = rsqrtf(s2 * (1.f / 64.f) - mean * mean + LN_EPS);
#pragma unroll
  for (int d = 0; d < 64; ++d) {
    float v = (vals[d] - mean) * rstd * on_g[d] + on_b[d];
    vals[d] = v * (float)ogate[(size_t)p * 64 + d];
  }
  int base = (lane >> 4) << 3;
  floatx8 acc[4] = {};
  for (int k0 = 0; k0 < 64; k0 += 32) {
    bf16x16 a;
#pragma unroll
    for (int h = 0; h < 8; ++h) {
      a[h] = (__bf16)vals[k0 + base + h];
      a[h + 8] = (__bf16)vals[k0 + base + h + 16];
    }
#pragma unroll
    for (int s4 = 0; s4 < 4; ++s4) {
      bf16x16 b = load_b_frag(Wo, 64, k0, s4 * 16, lane);
      acc[s4] = wmma_bf16(a, b, acc[s4]);
    }
  }
  int col = lane & 15, rowhi = (lane >> 4) << 3;
#pragma unroll
  for (int s4 = 0; s4 < 4; ++s4)
#pragma unroll
    for (int r = 0; r < 8; ++r) {
      int m = p0 + rowhi + r, n = s4 * 16 + col;
      size_t off = (size_t)m * 64 + n;
      edge[off] = edge[off] + bo[n] + acc[s4][r];
    }
}

// =============================================================== host dispatch
// d_in leaf order (setup_inputs dict order, depth-first):
enum {
  IN_NODE = 0, IN_EDGE = 1,
  A_NN_G = 2, A_NN_B = 3, A_EN_G = 4, A_EN_B = 5,
  A_WQ = 6, A_WK = 7, A_WV = 8, A_WB = 9, A_WG_W = 10, A_WG_B = 11,
  A_OUT_W = 12, A_OUT_B = 13,
  NT_LN_G = 14, NT_LN_B = 15, NT_W1_W = 16, NT_W1_B = 17, NT_W2_W = 18, NT_W2_B = 19,
  N2E_LN_G = 20, N2E_LN_B = 21, N2E_P_W = 22, N2E_P_B = 23, N2E_O_W = 24, N2E_O_B = 25,
  TO_BASE = 26, TI_BASE = 42,
  ET_LN_G = 58, ET_LN_B = 59, ET_W1_W = 60, ET_W1_B = 61, ET_W2_W = 62, ET_W2_B = 63
};

// bf16 weight-pool element offsets
constexpr size_t WQ = 0, WK = 32768, WV = 65536, WGW = 98304, WO = 131072,
                 NTW1 = 163840, NTW2 = 196608, PROJ = 229376, N2EO = 262144,
                 TOW = 278528, TIW = 303104, ETW1 = 327680, ETW2 = 335872,
                 WPOOL_ELEMS = 344064;

extern "C" void kernel_launch(void* const* d_in, const int* in_sizes, int n_in,
                              void* d_out, int out_size, void* d_ws, size_t ws_size,
                              hipStream_t stream) {
  (void)in_sizes; (void)n_in; (void)out_size; (void)ws_size;
  auto F = [&](int i) { return (const float*)d_in[i]; };

  float* node_out = (float*)d_out;                 // [512,128] final node
  float* edge_out = (float*)d_out + 512 * 128;     // [512,512,64] running edge

  char* ws = (char*)d_ws;
  size_t wo = 0;
  auto alloc = [&](size_t bytes) -> char* {
    char* r = ws + wo; wo = (wo + bytes + 255) & ~(size_t)255; return r;
  };
  __bf16* wpool    = (__bf16*)alloc(WPOOL_ELEMS * 2);
  __bf16* xnh_node = (__bf16*)alloc(512 * 128 * 2);
  __bf16* qb       = (__bf16*)alloc(512 * 256 * 2);
  __bf16* kb       = (__bf16*)alloc(512 * 256 * 2);
  __bf16* vb       = (__bf16*)alloc(512 * 256 * 2);
  __bf16* kTb      = (__bf16*)alloc(512 * 256 * 2);
  float*  gbuf     = (float*) alloc(512 * 256 * 4);
  float*  logits   = (float*) alloc((size_t)8 * EP * 4);
  __bf16* attnb    = (__bf16*)alloc((size_t)8 * EP * 2);
  __bf16* resg     = (__bf16*)alloc(512 * 256 * 2);
  float*  node_mid = (float*) alloc(512 * 128 * 4);
  __bf16* nln      = (__bf16*)alloc(512 * 128 * 2);
  __bf16* tbuf     = (__bf16*)alloc(512 * 256 * 2);
  float*  sbuf     = (float*) alloc(512 * 256 * 4);
  __bf16* EB0      = (__bf16*)alloc((size_t)EP * 64 * 2);  // tri xn -> tri mm out
  __bf16* EB1      = (__bf16*)alloc((size_t)EP * 64 * 2);  // Abuf   -> etrans xn
  __bf16* EB2      = (__bf16*)alloc((size_t)EP * 64 * 2);  // Bbuf   -> etrans h (lo)
  __bf16* EB3      = (__bf16*)alloc((size_t)EP * 64 * 2);  // ogate  -> etrans h (hi)

  auto pack = [&](int src, size_t off, int n) {
    k_pack<<<(n + 255) / 256, 256, 0, stream>>>(F(src), wpool + off, n);
  };
  pack(A_WQ, WQ, 32768);     pack(A_WK, WK, 32768);   pack(A_WV, WV, 32768);
  pack(A_WG_W, WGW, 32768);  pack(A_OUT_W, WO, 32768);
  pack(NT_W1_W, NTW1, 32768); pack(NT_W2_W, NTW2, 32768);
  pack(N2E_P_W, PROJ, 32768); pack(N2E_O_W, N2EO, 16384);
  for (int t = 0; t < 2; ++t) {
    int base = t ? TI_BASE : TO_BASE; size_t w = t ? TIW : TOW;
    pack(base + 2, w + 0, 4096);      // lp.w
    pack(base + 4, w + 4096, 4096);   // rp.w
    pack(base + 6, w + 8192, 4096);   // lg.w
    pack(base + 8, w + 12288, 4096);  // rg.w
    pack(base + 10, w + 16384, 4096); // og.w
    pack(base + 14, w + 20480, 4096); // out.w
  }
  pack(ET_W1_W, ETW1, 8192); pack(ET_W2_W, ETW2, 8192);

  // ---------------- node attention ----------------
  k_ln<<<512 / 8, 256, 0, stream>>>(F(IN_NODE), F(A_NN_G), F(A_NN_B), xnh_node, 128);
  k_gemm<<<dim3(32, 4), 32, 0, stream>>>(xnh_node, wpool + WQ, nullptr, nullptr,
                                         nullptr, qb, 256, 128, 0);
  k_gemm<<<dim3(32, 4), 32, 0, stream>>>(xnh_node, wpool + WK, nullptr, nullptr,
                                         nullptr, kb, 256, 128, 0);
  k_gemm<<<dim3(32, 4), 32, 0, stream>>>(xnh_node, wpool + WV, nullptr, nullptr,
                                         nullptr, vb, 256, 128, 0);
  k_gemm<<<dim3(32, 4), 32, 0, stream>>>(xnh_node, wpool + WGW, F(A_WG_B), nullptr,
                                         gbuf, nullptr, 256, 128, 2);
  k_transpose_k<<<512, 256, 0, stream>>>(kb, kTb);
  k_bias<<<EP / 8, 256, 0, stream>>>(F(IN_EDGE), F(A_EN_G), F(A_EN_B), F(A_WB), logits);
  k_logits<<<dim3(32, 8, 8), 32, 0, stream>>>(qb, kTb, logits);
  k_softmax<<<8 * 512 / 8, 256, 0, stream>>>(logits, attnb);
  k_av<<<dim3(32, 8), 32, 0, stream>>>(attnb, vb, gbuf, resg);
  k_gemm<<<dim3(32, 2), 32, 0, stream>>>(resg, wpool + WO, F(A_OUT_B), F(IN_NODE),
                                         node_mid, nullptr, 128, 256, 0);
  // ---------------- node transition ----------------
  k_ln<<<512 / 8, 256, 0, stream>>>(node_mid, F(NT_LN_G), F(NT_LN_B), nln, 128);
  k_gemm<<<dim3(32, 4), 32, 0, stream>>>(nln, wpool + NTW1, F(NT_W1_B), nullptr,
                                         nullptr, tbuf, 256, 128, 1);
  k_gemm<<<dim3(32, 2), 32, 0, stream>>>(tbuf, wpool + NTW2, F(NT_W2_B), node_mid,
                                         node_out, nullptr, 128, 256, 0);
  // ---------------- node -> edge ----------------
  k_ln<<<512 / 8, 256, 0, stream>>>(node_out, F(N2E_LN_G), F(N2E_LN_B), nln, 128);
  k_gemm<<<dim3(32, 4), 32, 0, stream>>>(nln, wpool + PROJ, F(N2E_P_B), nullptr,
                                         sbuf, nullptr, 256, 128, 0);
  k_n2e<<<dim3(32, 512), 32, 0, stream>>>(sbuf, wpool + N2EO, F(N2E_O_B),
                                          F(IN_EDGE), edge_out);
  // ---------------- triangle multiplications ----------------
  for (int t = 0; t < 2; ++t) {
    int base = t ? TI_BASE : TO_BASE;
    const __bf16* w = wpool + (t ? TIW : TOW);
    const __bf16 *Wlp = w, *Wrp = w + 4096, *Wlg = w + 8192, *Wrg = w + 12288,
                 *Wog = w + 16384, *Wow = w + 20480;
    k_ln<<<EP / 8, 256, 0, stream>>>(edge_out, F(base + 0), F(base + 1), EB0, 64);
    if (t == 0) {  // 'bikd,bjkd->bijd': A = left natural, B = right transposed
      k_gated_gemm<<<EP / 16, 32, 0, stream>>>(EB0, Wlp, F(base + 3), Wlg, F(base + 7), EB1, 0);
      k_gated_gemm<<<EP / 16, 32, 0, stream>>>(EB0, Wrp, F(base + 5), Wrg, F(base + 9), EB2, 1);
    } else {       // 'bkjd,bkid->bijd': A = right transposed, B = left natural
      k_gated_gemm<<<EP / 16, 32, 0, stream>>>(EB0, Wrp, F(base + 5), Wrg, F(base + 9), EB1, 1);
      k_gated_gemm<<<EP / 16, 32, 0, stream>>>(EB0, Wlp, F(base + 3), Wlg, F(base + 7), EB2, 0);
    }
    k_gemm<<<dim3(EP / 16, 1), 32, 0, stream>>>(EB0, Wog, F(base + 11), nullptr,
                                                nullptr, EB3, 64, 64, 2);
    k_tri_mm<<<dim3(16, 8, 64), 32, 0, stream>>>(EB1, EB2, EB0);
    k_tri_post<<<EP / 16, 32, 0, stream>>>(EB0, F(base + 12), F(base + 13), EB3,
                                           Wow, F(base + 15), edge_out);
  }
  // ---------------- edge transition ----------------
  k_ln<<<EP / 8, 256, 0, stream>>>(edge_out, F(ET_LN_G), F(ET_LN_B), EB1, 64);
  k_gemm<<<dim3(EP / 16, 2), 32, 0, stream>>>(EB1, wpool + ETW1, F(ET_W1_B), nullptr,
                                              nullptr, EB2, 128, 64, 1);
  k_gemm<<<dim3(EP / 16, 1), 32, 0, stream>>>(EB2, wpool + ETW2, F(ET_W2_B), edge_out,
                                              edge_out, nullptr, 64, 128, 0);
}
